// LeafBlockAttention_68307159876022
// MI455X (gfx1250) — compile-verified, matrix operand order checked by reading, not run on hardware
//
#include <hip/hip_runtime.h>
#include <hip/hip_bf16.h>

// MI455X / gfx1250 implementation. All matrix math uses v_wmma_f32_16x16x32_f16
// (wave32 WMMA, f16 inputs, f32 accumulate). Every WMMA operand is fed by
// 2x ds_load_b128 (CDNA5 16-bit A-layout = two contiguous 8-half groups/lane);
// all B operands are staged transposed in LDS so no column gathers remain.
//
// Workspace layout (bytes):
//   [0, 100663296)                : qkv    f16 [65536][768]
//   [100663296, 101449728)        : kvnode f16 [512][768]   (block-node qkv = mean)
//   [101449728, 135004160)        : oat    f16 [65536][256] (attention output)
// Requires ws_size >= ~135 MB.

#define Bv 4
#define NBv 128
#define BSv 128
#define Cv 256
#define Hv 8
#define HD 32
#define NROWS (Bv * NBv * BSv)   // 65536

typedef __attribute__((ext_vector_type(16))) _Float16 v16h;
typedef __attribute__((ext_vector_type(8)))  _Float16 v8h;
typedef __attribute__((ext_vector_type(4)))  _Float16 v4h;
typedef __attribute__((ext_vector_type(8)))  float    v8f;

// ---------------------------------------------------------------------------
// CDNA5 16-bit fragment loader (ISA 7.12.2 layout, wave32).
// A-matrix 16x32 (MxK): lane 0-15 -> M = lane, lanes 16-31 -> same M, K+8.
// Element e: k = k0 + e + (e>=8 ? 8 : 0) + (lane>=16 ? 8 : 0), i.e. two
// contiguous 8-half groups at kbase and kbase+16 -> 2x ds_load_b128.
// Requires ld % 8 == 0 and 16-byte aligned base.
// ---------------------------------------------------------------------------
__device__ __forceinline__ v16h load_frag_row(const _Float16* p, int ld, int r0, int k0) {
  const int lane  = threadIdx.x & 31;
  const int row   = r0 + (lane & 15);
  const int kbase = k0 + ((lane & 16) ? 8 : 0);
  const _Float16* q = p + row * ld + kbase;
  v8h lo = *(const v8h*)(q);
  v8h hi = *(const v8h*)(q + 16);
  return __builtin_shufflevector(lo, hi, 0, 1, 2, 3, 4, 5, 6, 7,
                                 8, 9, 10, 11, 12, 13, 14, 15);
}

__device__ __forceinline__ v8f wmma_f16(v16h a, v16h b, v8f c) {
  return __builtin_amdgcn_wmma_f32_16x16x32_f16(false, a, false, b, (short)0, c,
                                                false, false);
}

// ---------------------------------------------------------------------------
// Kernel 1: QKV GEMM.  y[65536][768] (f16) = x[65536][256] @ w[256][768] + b
// grid (NROWS/64, 768/64), block 128 (4 waves). 64x64 C tile per workgroup.
// Weight tile staged TRANSPOSED (Bt[n][k]) so B fragments are row-layout.
// ---------------------------------------------------------------------------
__global__ __launch_bounds__(128)
void qkv_gemm(const float* __restrict__ x, const float* __restrict__ w,
              const float* __restrict__ bias, _Float16* __restrict__ y) {
  __shared__ alignas(16) _Float16 As[64][40];
  __shared__ alignas(16) _Float16 Bt[64][40];   // Bt[n][k]
  const int m_base = blockIdx.x * 64;
  const int n_base = blockIdx.y * 64;
  const int tid  = threadIdx.x;
  const int wave = tid >> 5;

  v8f acc[4] = {};
  for (int kc = 0; kc < Cv; kc += 32) {
    // stage A: 64x32 f32 -> f16, float4 loads + b64 LDS stores
    for (int i = tid; i < 64 * 8; i += 128) {
      int r = i >> 3, c4 = (i & 7) * 4;
      float4 xv = *(const float4*)(x + (size_t)(m_base + r) * Cv + kc + c4);
      v4h o = {(_Float16)xv.x, (_Float16)xv.y, (_Float16)xv.z, (_Float16)xv.w};
      *(v4h*)&As[r][c4] = o;
    }
    // stage B transposed: Bt[c][r] = w[kc+r][n_base+c]
    for (int i = tid; i < 32 * 16; i += 128) {
      int r = i >> 4, c4 = (i & 15) * 4;
      float4 wv = *(const float4*)(w + (size_t)(kc + r) * 768 + n_base + c4);
      Bt[c4 + 0][r] = (_Float16)wv.x;
      Bt[c4 + 1][r] = (_Float16)wv.y;
      Bt[c4 + 2][r] = (_Float16)wv.z;
      Bt[c4 + 3][r] = (_Float16)wv.w;
    }
    __syncthreads();
    v16h a = load_frag_row(&As[0][0], 40, wave * 16, 0);
#pragma unroll
    for (int nt = 0; nt < 4; ++nt) {
      v16h b = load_frag_row(&Bt[0][0], 40, nt * 16, 0);
      acc[nt] = wmma_f16(a, b, acc[nt]);
    }
    __syncthreads();
  }

  const int lane = tid & 31, hi = lane >> 4, nl = lane & 15;
#pragma unroll
  for (int nt = 0; nt < 4; ++nt) {
    int n = n_base + nt * 16 + nl;
    float bv = bias[n];
#pragma unroll
    for (int r = 0; r < 8; ++r) {
      int m = m_base + wave * 16 + r + 8 * hi;
      y[(size_t)m * 768 + n] = (_Float16)(acc[nt][r] + bv);
    }
  }
}

// ---------------------------------------------------------------------------
// Kernel 2: block-node qkv row = mean over the 128 rows of each block.
// grid 512, block 128 (threads 0..95 each own 8 contiguous columns).
// ---------------------------------------------------------------------------
__global__ __launch_bounds__(128)
void block_mean(const _Float16* __restrict__ qkv, _Float16* __restrict__ kvnode) {
  const int blk = blockIdx.x;                 // b*128 + nb
  const int t = threadIdx.x;
  if (t >= 96) return;
  const size_t base = (size_t)blk * BSv * 768 + t * 8;
  float s[8] = {0, 0, 0, 0, 0, 0, 0, 0};
  for (int r = 0; r < BSv; ++r) {
    v8h v = *(const v8h*)(qkv + base + (size_t)r * 768);
#pragma unroll
    for (int j = 0; j < 8; ++j) s[j] += (float)v[j];
  }
  v8h o;
#pragma unroll
  for (int j = 0; j < 8; ++j) o[j] = (_Float16)(s[j] * (1.0f / 128.0f));
  *(v8h*)(kvnode + (size_t)blk * 768 + t * 8) = o;
}

// ---------------------------------------------------------------------------
// Kernel 3: fused per-(batch, block, head, query-half) attention.
//   scores = (q*scale) @ k^T          (WMMA, K = hd = 32)
//   scores += bias3; mask; softmax; += gate (edge_feats @ w_gate)
//   out    = combined @ v             (WMMA, K padded 129 -> 160)
// grid 8192, block 128 (4 waves; wave w owns query rows [16w, 16w+16)).
// V is staged TRANSPOSED (vst[d][k]) so its fragments are row-layout b128.
// ---------------------------------------------------------------------------
__global__ __launch_bounds__(128)
void attn_kernel(const _Float16* __restrict__ qkv, const _Float16* __restrict__ kvnode,
                 const int* __restrict__ amask, const float* __restrict__ ef,
                 const float* __restrict__ w_gate, const float* __restrict__ b_gate,
                 _Float16* __restrict__ oat) {
  __shared__ alignas(16) _Float16 qs[64][40];    // q head slice (pre-scaled)
  __shared__ alignas(16) _Float16 ks[144][40];   // k rows, zero-padded to 144
  __shared__ alignas(16) _Float16 vst[32][168];  // v TRANSPOSED: vst[d][k], k zero-padded
  __shared__ alignas(16) _Float16 sc[64][168];   // scores -> probs -> combined

  const int flat = blockIdx.x;        // ((b*NB + nb)*H + h)*2 + half
  const int half = flat & 1;
  const int h    = (flat >> 1) & 7;
  const int nb   = (flat >> 4) & 127;
  const int bq   = flat >> 11;
  const int tid  = threadIdx.x;
  const int wave = tid >> 5;

  const size_t rowbase  = (size_t)bq * (NBv * BSv) + (size_t)nb * BSv;
  const size_t nodebase = ((size_t)bq * NBv + nb) * 768;
  const float  scale    = 0.17677669529663689f;   // hd^-0.5

  // ---- stage q/k/v head slices into LDS via b128 chunks ----
  for (int i = tid; i < 64 * 4; i += 128) {          // q: 64 rows x 4 chunks
    int r = i >> 2, d8 = (i & 3) * 8;
    v8h v = *(const v8h*)(qkv + (rowbase + half * 64 + r) * 768 + h * HD + d8);
    v8h o;
#pragma unroll
    for (int j = 0; j < 8; ++j) o[j] = (_Float16)((float)v[j] * scale);
    *(v8h*)&qs[r][d8] = o;
  }
  for (int i = tid; i < 144 * 4; i += 128) {         // k: 144 rows x 4 chunks
    int r = i >> 2, d8 = (i & 3) * 8;
    v8h v = {};
    if (r < BSv)       v = *(const v8h*)(qkv + (rowbase + r) * 768 + Cv + h * HD + d8);
    else if (r == BSv) v = *(const v8h*)(kvnode + nodebase + Cv + h * HD + d8);
    *(v8h*)&ks[r][d8] = v;
  }
  for (int i = tid; i < 160 * 4; i += 128) {         // v: transpose while staging
    int r = i >> 2, d8 = (i & 3) * 8;                // r = key idx (K dim)
    v8h v = {};
    if (r < BSv)       v = *(const v8h*)(qkv + (rowbase + r) * 768 + 2 * Cv + h * HD + d8);
    else if (r == BSv) v = *(const v8h*)(kvnode + nodebase + 2 * Cv + h * HD + d8);
#pragma unroll
    for (int j = 0; j < 8; ++j) vst[d8 + j][r] = v[j];
  }
  __syncthreads();

  // ---- scores: q @ k^T (k^T column-major == A-style load of k rows) ----
  {
    v8f acc[9] = {};
    v16h a = load_frag_row(&qs[0][0], 40, wave * 16, 0);
#pragma unroll
    for (int nt = 0; nt < 9; ++nt) {
      v16h b = load_frag_row(&ks[0][0], 40, nt * 16, 0);
      acc[nt] = wmma_f16(a, b, acc[nt]);
    }
    const int lane = tid & 31, hi = lane >> 4, nl = lane & 15;
#pragma unroll
    for (int nt = 0; nt < 9; ++nt)
#pragma unroll
      for (int r = 0; r < 8; ++r)
        sc[wave * 16 + r + 8 * hi][nt * 16 + nl] = (_Float16)acc[nt][r];
  }
  __syncthreads();

  // ---- bias + mask + softmax + gate (one thread per query row) ----
  if (tid < 64) {
    const int q  = tid;
    const int qg = half * 64 + q;                         // query idx in block
    const float* efq = ef + ((size_t)nb * BSv + qg) * (BSv + 1) * 4;
    const int*   mq  = amask + ((size_t)nb * BSv + qg) * (BSv + 1);
    __builtin_prefetch(mq, 0, 1);                         // global_prefetch_b8
    __builtin_prefetch(efq, 0, 1);

    float mx = -3.0e38f;
    for (int k = 0; k <= BSv; ++k) {
      bool special = (k == BSv) || (k == qg);
      float s = (float)sc[q][k] + (special ? 1.0f : efq[k * 4 + 3]);
      if (mq[k] == 0) s = -30000.0f;
      sc[q][k] = (_Float16)s;
      mx = fmaxf(mx, s);
    }
    float sum = 0.0f;
    for (int k = 0; k <= BSv; ++k) {
      float p = __expf((float)sc[q][k] - mx);
      sum += p;
      sc[q][k] = (_Float16)p;
    }
    const float inv = 1.0f / sum;
    const float wg0 = w_gate[0 * Hv + h], wg1 = w_gate[1 * Hv + h];
    const float wg2 = w_gate[2 * Hv + h], wg3 = w_gate[3 * Hv + h];
    const float bg  = b_gate[h];
    for (int k = 0; k <= BSv; ++k) {
      float gate = 0.0f;
      if (mq[k] != 0) {
        bool special = (k == BSv) || (k == qg);
        float e0 = special ? 0.0f : efq[k * 4 + 0];
        float e1 = special ? 0.0f : efq[k * 4 + 1];
        float e2 = special ? 0.0f : efq[k * 4 + 2];
        float e3 = special ? 1.0f : efq[k * 4 + 3];
        gate = e0 * wg0 + e1 * wg1 + e2 * wg2 + e3 * wg3 + bg;
      }
      sc[q][k] = (_Float16)((float)sc[q][k] * inv + gate);
    }
    for (int k = BSv + 1; k < 160; ++k) sc[q][k] = (_Float16)0.0f;
  }
  __syncthreads();

  // ---- out = combined @ v  (K padded to 160 -> 5 clean chunks of 32) ----
  {
    v8f acc[2] = {};
#pragma unroll
    for (int kc = 0; kc < 160; kc += 32) {
      v16h a = load_frag_row(&sc[0][0], 168, wave * 16, kc);
#pragma unroll
      for (int nt = 0; nt < 2; ++nt) {
        v16h b = load_frag_row(&vst[0][0], 168, nt * 16, kc);  // B[k][n] = vst[n][k]
        acc[nt] = wmma_f16(a, b, acc[nt]);
      }
    }
    const int lane = tid & 31, hi = lane >> 4, nl = lane & 15;
#pragma unroll
    for (int nt = 0; nt < 2; ++nt)
#pragma unroll
      for (int r = 0; r < 8; ++r) {
        int m = half * 64 + wave * 16 + r + 8 * hi;
        oat[(rowbase + m) * Cv + h * HD + nt * 16 + nl] = (_Float16)acc[nt][r];
      }
  }
}

// ---------------------------------------------------------------------------
// Kernel 4: output projection.  out[65536][256] (f32) = oat @ w_proj + b_proj
// grid (NROWS/64, 256/64), block 128.  Weight tile staged transposed.
// ---------------------------------------------------------------------------
__global__ __launch_bounds__(128)
void proj_gemm(const _Float16* __restrict__ a, const float* __restrict__ w,
               const float* __restrict__ bias, float* __restrict__ out) {
  __shared__ alignas(16) _Float16 As[64][40];
  __shared__ alignas(16) _Float16 Bt[64][40];   // Bt[n][k]
  const int m_base = blockIdx.x * 64;
  const int n_base = blockIdx.y * 64;
  const int tid  = threadIdx.x;
  const int wave = tid >> 5;

  v8f acc[4] = {};
  for (int kc = 0; kc < Cv; kc += 32) {
    // stage A (already f16): b128 copies
    for (int i = tid; i < 64 * 4; i += 128) {
      int r = i >> 2, d8 = (i & 3) * 8;
      *(v8h*)&As[r][d8] = *(const v8h*)(a + (size_t)(m_base + r) * Cv + kc + d8);
    }
    // stage B transposed: Bt[c][r] = w[kc+r][n_base+c]
    for (int i = tid; i < 32 * 16; i += 128) {
      int r = i >> 4, c4 = (i & 15) * 4;
      float4 wv = *(const float4*)(w + (size_t)(kc + r) * Cv + n_base + c4);
      Bt[c4 + 0][r] = (_Float16)wv.x;
      Bt[c4 + 1][r] = (_Float16)wv.y;
      Bt[c4 + 2][r] = (_Float16)wv.z;
      Bt[c4 + 3][r] = (_Float16)wv.w;
    }
    __syncthreads();
    v16h afr = load_frag_row(&As[0][0], 40, wave * 16, 0);
#pragma unroll
    for (int nt = 0; nt < 4; ++nt) {
      v16h bfr = load_frag_row(&Bt[0][0], 40, nt * 16, 0);
      acc[nt] = wmma_f16(afr, bfr, acc[nt]);
    }
    __syncthreads();
  }

  const int lane = tid & 31, hi = lane >> 4, nl = lane & 15;
#pragma unroll
  for (int nt = 0; nt < 4; ++nt) {
    int n = n_base + nt * 16 + nl;
    float bv = bias[n];
#pragma unroll
    for (int r = 0; r < 8; ++r) {
      int m = m_base + wave * 16 + r + 8 * hi;
      out[(size_t)m * Cv + n] = acc[nt][r] + bv;
    }
  }
}

// ---------------------------------------------------------------------------
extern "C" void kernel_launch(void* const* d_in, const int* in_sizes, int n_in,
                              void* d_out, int out_size, void* d_ws, size_t ws_size,
                              hipStream_t stream) {
  const float* x      = (const float*)d_in[0];
  const int*   amask  = (const int*)d_in[1];
  const float* ef     = (const float*)d_in[2];
  const float* w_qkv  = (const float*)d_in[3];
  const float* b_qkv  = (const float*)d_in[4];
  const float* w_proj = (const float*)d_in[5];
  const float* b_proj = (const float*)d_in[6];
  const float* w_gate = (const float*)d_in[7];
  const float* b_gate = (const float*)d_in[8];
  float* out = (float*)d_out;

  char* ws = (char*)d_ws;
  _Float16* qkv    = (_Float16*)ws;                                   // 100.7 MB
  _Float16* kvnode = (_Float16*)(ws + (size_t)NROWS * 768 * 2);       // 0.79 MB
  _Float16* oat    = (_Float16*)(ws + (size_t)NROWS * 768 * 2 + (size_t)512 * 768 * 2);

  qkv_gemm<<<dim3(NROWS / 64, 768 / 64), 128, 0, stream>>>(x, w_qkv, b_qkv, qkv);
  block_mean<<<Bv * NBv, 128, 0, stream>>>(qkv, kvnode);
  attn_kernel<<<Bv * NBv * Hv * 2, 128, 0, stream>>>(qkv, kvnode, amask, ef,
                                                     w_gate, b_gate, oat);
  proj_gemm<<<dim3(NROWS / 64, Cv / 64), 128, 0, stream>>>(oat, w_proj, b_proj, out);
}
